// BloomSelfAttentionWrapper_56839597195737
// MI455X (gfx1250) — compile-verified, matrix operand order checked by reading, not run on hardware
//
#include <hip/hip_runtime.h>
#include <math.h>

typedef __attribute__((ext_vector_type(16))) _Float16 v16h;
typedef __attribute__((ext_vector_type(8)))  _Float16 v8h;
typedef __attribute__((ext_vector_type(8)))  float    v8f;

static constexpr int SEQ   = 256;
static constexpr int HIDN  = 1024;
static constexpr int NHEAD = 16;
static constexpr int HD    = 64;
static constexpr int NQKV  = 3072;   // 3 * HIDN
#define INV_NORM   0.125f            // 1/sqrt(64)
#define INV_SQRT_D 0.125f

#define WMMA_F16(a, b, c) \
  __builtin_amdgcn_wmma_f32_16x16x32_f16(false, (a), false, (b), (short)0, (c), false, false)

// ---- f16 fragment loader (CDNA5 16-bit A/B layout, ISA 7.12.2) -------------
// lane<16 : row=lane,    K = k0+{0..7} and k0+{16..23}
// lane>=16: row=lane-16, K = k0+{8..15} and k0+{24..31}
// Works for A (lane = M) and for B when B^T is row-major (lane = N).
__device__ inline v16h frag_f16(const _Float16* __restrict__ P, int ld,
                                int row0, int k0) {
  const int lane = threadIdx.x & 31;
  const int half = lane >> 4;
  const _Float16* base = P + (row0 + (lane & 15)) * ld + k0 + half * 8;
  const v8h lo = *(const v8h*)(base);        // 16B load
  const v8h hi = *(const v8h*)(base + 16);   // 16B load
  v16h r;
#pragma unroll
  for (int t = 0; t < 8; ++t) { r[t] = lo[t]; r[8 + t] = hi[t]; }
  return r;
}

// ---- Pack: f32 -> f16 flat copy --------------------------------------------
__global__ void k_pack_f16(const float* __restrict__ in, _Float16* __restrict__ out, int n) {
  for (int i = blockIdx.x * blockDim.x + threadIdx.x; i < n; i += gridDim.x * blockDim.x)
    out[i] = (_Float16)in[i];
}

// ---- Pack: W [K x N] f32 -> WT [N x K] f16, LDS-tiled transpose ------------
__global__ void k_packT(const float* __restrict__ W, _Float16* __restrict__ WT,
                        int K, int N) {
  __shared__ float tile[32][33];            // +1 pad: no bank conflicts
  const int nb = blockIdx.x * 32;
  const int kb = blockIdx.y * 32;
#pragma unroll
  for (int r = 0; r < 4; ++r)
    tile[threadIdx.y + 8 * r][threadIdx.x] =
        W[(kb + threadIdx.y + 8 * r) * N + nb + threadIdx.x];
  __syncthreads();
#pragma unroll
  for (int r = 0; r < 4; ++r)
    WT[(nb + threadIdx.y + 8 * r) * K + kb + threadIdx.x] =
        (_Float16)tile[threadIdx.x][threadIdx.y + 8 * r];
}

// ---- Kernel: fused QKV projection, 16x64 tile per wave ---------------------
__global__ void k_qkv(const _Float16* __restrict__ hsH, const _Float16* __restrict__ WqkvT,
                      const float* __restrict__ bqkv, float* __restrict__ fused) {
  const int n0 = blockIdx.x * 64;
  const int m0 = blockIdx.y * 16;
  v8f acc[4] = {};
  for (int k0 = 0; k0 < HIDN; k0 += 32) {
    const v16h a = frag_f16(hsH, HIDN, m0, k0);
#pragma unroll
    for (int c = 0; c < 4; ++c) {
      const v16h b = frag_f16(WqkvT, HIDN, n0 + c * 16, k0);
      acc[c] = WMMA_F16(a, b, acc[c]);
    }
  }
  const int lane = threadIdx.x & 31;
  const int half = lane >> 4;
#pragma unroll
  for (int c = 0; c < 4; ++c) {
    const int col = n0 + c * 16 + (lane & 15);
    const float bias = bqkv[col];
#pragma unroll
    for (int r = 0; r < 8; ++r)
      fused[(m0 + half * 8 + r) * NQKV + col] = acc[c][r] + bias;
  }
}

// ---- Pack per-head operands: qh/kh [h][s][d] f16, vT/kT [h][d][s] ----------
__global__ void k_pack_heads(const float* __restrict__ fused, _Float16* __restrict__ qh,
                             _Float16* __restrict__ kh, _Float16* __restrict__ vTh,
                             float* __restrict__ kT) {
  const int h = blockIdx.x;
  for (int idx = threadIdx.x; idx < SEQ * HD; idx += blockDim.x) {
    const int j = idx >> 6, d = idx & 63;
    const float qv = fused[j * NQKV + h * 192 + d];
    const float kv = fused[j * NQKV + h * 192 + HD + d];
    const float vv = fused[j * NQKV + h * 192 + 2 * HD + d];
    qh[(h * SEQ + j) * HD + d]  = (_Float16)qv;
    kh[(h * SEQ + j) * HD + d]  = (_Float16)kv;
    vTh[(h * HD + d) * SEQ + j] = (_Float16)vv;
    kT[(h * HD + d) * SEQ + j]  = kv;
  }
}

// ---- Kernel: scores = alibi + inv_norm * Q.K^T + mask, 16x64 tile ----------
__global__ void k_scores(const _Float16* __restrict__ qh, const _Float16* __restrict__ kh,
                         const float* __restrict__ alibi, const float* __restrict__ mask,
                         float* __restrict__ scores) {
  const int j0 = blockIdx.x * 64;
  const int i0 = blockIdx.y * 16;
  const int h  = blockIdx.z;
  const _Float16* Q = qh + h * SEQ * HD;   // ld = HD
  const _Float16* K = kh + h * SEQ * HD;   // lane acts as N=j, K-dim contiguous
  v8f acc[4] = {};
  for (int k0 = 0; k0 < HD; k0 += 32) {
    const v16h a = frag_f16(Q, HD, i0, k0);
#pragma unroll
    for (int c = 0; c < 4; ++c) {
      const v16h b = frag_f16(K, HD, j0 + c * 16, k0);
      acc[c] = WMMA_F16(a, b, acc[c]);
    }
  }
  const int lane = threadIdx.x & 31;
  const int half = lane >> 4;
#pragma unroll
  for (int c = 0; c < 4; ++c) {
    const int j = j0 + c * 16 + (lane & 15);
    const float ab = alibi[h * SEQ + j];
#pragma unroll
    for (int r = 0; r < 8; ++r) {
      const int i = i0 + half * 8 + r;
      scores[(h * SEQ + i) * SEQ + j] = ab + INV_NORM * acc[c][r] + mask[i * SEQ + j];
    }
  }
}

// ---- Kernel: row softmax. Rows staged to LDS via CDNA5 async-to-LDS path ---
// One wave32 per row; each wave stages its 1KB row with two
// global_load_async_to_lds_b128 issues (INST_OFFSET applies to both the
// global and the LDS address, ISA 08_async_tensor §4.4), then waits on
// ASYNCcnt before the shuffle reductions.
__global__ void k_softmax(const float* __restrict__ scores, float* __restrict__ p,
                          _Float16* __restrict__ ph) {
  __shared__ float lrow[8 * SEQ];
  const int lane = threadIdx.x & 31;
  const int wave = threadIdx.x >> 5;
  const int row  = blockIdx.x * 8 + wave;

  const unsigned ldsb = (unsigned)(size_t)&lrow[wave * SEQ + lane * 4];
  const unsigned gb   = (unsigned)((row * SEQ + lane * 4) * sizeof(float));
  asm volatile("global_load_async_to_lds_b128 %0, %1, %2\n\t"
               "global_load_async_to_lds_b128 %0, %1, %2 offset:512"
               :: "v"(ldsb), "v"(gb), "s"(scores) : "memory");
  asm volatile("s_wait_asynccnt 0" ::: "memory");

  const float* lr = &lrow[wave * SEQ];
  float v[8];
  float m = -3.4e38f;
#pragma unroll
  for (int t = 0; t < 8; ++t) { v[t] = lr[lane + t * 32]; m = fmaxf(m, v[t]); }
  for (int o = 16; o > 0; o >>= 1) m = fmaxf(m, __shfl_xor(m, o, 32));
  float sum = 0.f;
#pragma unroll
  for (int t = 0; t < 8; ++t) { v[t] = __expf(v[t] - m); sum += v[t]; }
  for (int o = 16; o > 0; o >>= 1) sum += __shfl_xor(sum, o, 32);
  const float inv = 1.0f / sum;
#pragma unroll
  for (int t = 0; t < 8; ++t) {
    const float pv = v[t] * inv;
    p[row * SEQ + lane + t * 32]  = pv;
    ph[row * SEQ + lane + t * 32] = (_Float16)pv;
  }
}

// ---- Kernel: zero importance accumulator -----------------------------------
__global__ void k_zero(float* __restrict__ imp) {
  const int t = blockIdx.x * blockDim.x + threadIdx.x;
  if (t < NHEAD * HD) imp[t] = 0.f;
}

// ---- Kernel: fused delta/p_diff softmax + importance reduction (fp32) ------
__global__ void k_importance(const float* __restrict__ fused, const float* __restrict__ kT,
                             const float* __restrict__ scores, const float* __restrict__ p,
                             const float* __restrict__ alibi, float* __restrict__ imp) {
  const int lane = threadIdx.x & 31;
  const int row  = blockIdx.x * (blockDim.x >> 5) + (threadIdx.x >> 5); // 0..262143
  const int h = row >> 14;
  const int d = (row >> 8) & 63;
  const int i = row & 255;
  const float qd = fused[i * NQKV + h * 192 + d];
  const float* srow = scores + (h * SEQ + i) * SEQ;
  const float* prow = p + (h * SEQ + i) * SEQ;
  const float* krow = kT + (h * HD + d) * SEQ;
  const float* arow = alibi + h * SEQ;
  float df[8];
  float m = -3.4e38f;
#pragma unroll
  for (int t = 0; t < 8; ++t) {
    const int j = lane + t * 32;
    const float delta = arow[j] + INV_NORM * qd * krow[j];
    const float x = (srow[j] - delta) * INV_SQRT_D;
    df[t] = x; m = fmaxf(m, x);
  }
  for (int o = 16; o > 0; o >>= 1) m = fmaxf(m, __shfl_xor(m, o, 32));
  float sum = 0.f;
#pragma unroll
  for (int t = 0; t < 8; ++t) { df[t] = __expf(df[t] - m); sum += df[t]; }
  for (int o = 16; o > 0; o >>= 1) sum += __shfl_xor(sum, o, 32);
  const float inv = 1.0f / sum;
  float acc = 0.f;
#pragma unroll
  for (int t = 0; t < 8; ++t) {
    const float e = df[t] * inv - prow[lane + t * 32];
    acc += e * e;
  }
  for (int o = 16; o > 0; o >>= 1) acc += __shfl_xor(acc, o, 32);
  if (lane == 0) atomicAdd(&imp[h * HD + d], acc);
}

// ---- Kernel: context = p @ V, 16x64 tile (full head dim) per wave ----------
__global__ void k_context(const _Float16* __restrict__ ph, const _Float16* __restrict__ vTh,
                          _Float16* __restrict__ ctxH) {
  const int i0 = blockIdx.x * 16;
  const int h  = blockIdx.y;
  const _Float16* P = ph + h * SEQ * SEQ;    // ld = SEQ
  const _Float16* V = vTh + h * HD * SEQ;    // lane acts as N=d, K=j contiguous
  v8f acc[4] = {};
  for (int k0 = 0; k0 < SEQ; k0 += 32) {
    const v16h a = frag_f16(P, SEQ, i0, k0);
#pragma unroll
    for (int c = 0; c < 4; ++c) {
      const v16h b = frag_f16(V, SEQ, c * 16, k0);
      acc[c] = WMMA_F16(a, b, acc[c]);
    }
  }
  const int lane = threadIdx.x & 31;
  const int half = lane >> 4;
#pragma unroll
  for (int c = 0; c < 4; ++c) {
    const int col = h * HD + c * 16 + (lane & 15);
#pragma unroll
    for (int r = 0; r < 8; ++r)
      ctxH[(i0 + half * 8 + r) * HIDN + col] = (_Float16)acc[c][r];
  }
}

// ---- Kernel: out = ctx @ W_dense + b + residual, 16x64 tile ----------------
__global__ void k_dense(const _Float16* __restrict__ ctxH, const _Float16* __restrict__ WdT,
                        const float* __restrict__ bd, const float* __restrict__ resid,
                        float* __restrict__ out) {
  const int n0 = blockIdx.x * 64;
  const int m0 = blockIdx.y * 16;
  v8f acc[4] = {};
  for (int k0 = 0; k0 < HIDN; k0 += 32) {
    const v16h a = frag_f16(ctxH, HIDN, m0, k0);
#pragma unroll
    for (int c = 0; c < 4; ++c) {
      const v16h b = frag_f16(WdT, HIDN, n0 + c * 16, k0);
      acc[c] = WMMA_F16(a, b, acc[c]);
    }
  }
  const int lane = threadIdx.x & 31;
  const int half = lane >> 4;
#pragma unroll
  for (int c = 0; c < 4; ++c) {
    const int col = n0 + c * 16 + (lane & 15);
    const float bias = bd[col];
#pragma unroll
    for (int r = 0; r < 8; ++r) {
      const int rowi = m0 + half * 8 + r;
      out[rowi * HIDN + col] = acc[c][r] + bias + resid[rowi * HIDN + col];
    }
  }
}

extern "C" void kernel_launch(void* const* d_in, const int* in_sizes, int n_in,
                              void* d_out, int out_size, void* d_ws, size_t ws_size,
                              hipStream_t stream) {
  (void)in_sizes; (void)n_in; (void)out_size; (void)ws_size;
  const float* hs    = (const float*)d_in[0];
  const float* resid = (const float*)d_in[1];
  const float* alibi = (const float*)d_in[2];
  const float* mask  = (const float*)d_in[3];
  const float* Wqkv  = (const float*)d_in[4];
  const float* bqkv  = (const float*)d_in[5];
  const float* Wd    = (const float*)d_in[6];
  const float* bd    = (const float*)d_in[7];
  float* out = (float*)d_out;
  float* imp = out + SEQ * HIDN;          // qk_importance follows `out`

  // Workspace layout (~26 MB)
  float* fused  = (float*)d_ws;                       // 786432 f32
  float* scores = fused + SEQ * NQKV;                 // 1048576
  float* p      = scores + NHEAD * SEQ * SEQ;         // 1048576
  float* kT     = p + NHEAD * SEQ * SEQ;              // 262144
  _Float16* hsH   = (_Float16*)(kT + NHEAD * HD * SEQ);
  _Float16* WqkvT = hsH + SEQ * HIDN;                 // 3145728 f16
  _Float16* WdT   = WqkvT + NQKV * HIDN;              // 1048576
  _Float16* qh    = WdT + HIDN * HIDN;                // 262144
  _Float16* kh    = qh + NHEAD * SEQ * HD;
  _Float16* vTh   = kh + NHEAD * SEQ * HD;
  _Float16* ph    = vTh + NHEAD * SEQ * HD;           // 1048576
  _Float16* ctxH  = ph + NHEAD * SEQ * SEQ;           // 262144

  // One-time f16 packs (L2-resident, bandwidth-trivial)
  k_pack_f16 <<<dim3(256), dim3(256), 0, stream>>>(hs, hsH, SEQ * HIDN);
  k_packT    <<<dim3(NQKV / 32, HIDN / 32), dim3(32, 8), 0, stream>>>(Wqkv, WqkvT, HIDN, NQKV);
  k_packT    <<<dim3(HIDN / 32, HIDN / 32), dim3(32, 8), 0, stream>>>(Wd, WdT, HIDN, HIDN);

  k_qkv       <<<dim3(NQKV / 64, SEQ / 16),       dim3(32),  0, stream>>>(hsH, WqkvT, bqkv, fused);
  k_pack_heads<<<dim3(NHEAD),                     dim3(256), 0, stream>>>(fused, qh, kh, vTh, kT);
  k_scores    <<<dim3(SEQ / 64, SEQ / 16, NHEAD), dim3(32),  0, stream>>>(qh, kh, alibi, mask, scores);
  k_softmax   <<<dim3(NHEAD * SEQ / 8),           dim3(256), 0, stream>>>(scores, p, ph);
  k_zero      <<<dim3(4),                         dim3(256), 0, stream>>>(imp);
  k_importance<<<dim3(NHEAD * HD * SEQ / 8),      dim3(256), 0, stream>>>(fused, kT, scores, p, alibi, imp);
  k_context   <<<dim3(SEQ / 16, NHEAD),           dim3(32),  0, stream>>>(ph, vTh, ctxH);
  k_dense     <<<dim3(HIDN / 64, SEQ / 16),       dim3(32),  0, stream>>>(ctxH, WdT, bd, resid, out);
}
